// MatcherDynamicK_87187836109449
// MI455X (gfx1250) — matrix operand order repaired and checked by reading, not compiled, not run on hardware
//
#include <hip/hip_runtime.h>
#include <hip/hip_bf16.h>

typedef __attribute__((ext_vector_type(16))) _Float16 v16h;
typedef __attribute__((ext_vector_type(8)))  float    v8f;
typedef int vs4i __attribute__((vector_size(16)));   // matches builtin param element type

namespace {
constexpr int BB = 16;     // batch
constexpr int NQ = 4096;   // queries
constexpr int CC = 80;     // classes
constexpr int CPAD = 96;   // classes padded to 3*32 for WMMA K
constexpr int MG = 256;    // gt boxes
constexpr int KOTA = 10;
constexpr int IMAX = 0x7fffffff;
}

__device__ __forceinline__ float fastrcp(float x) { return __builtin_amdgcn_rcpf(x); }

// ---------------------------------------------------------------------------
// CDNA5 async global->LDS staging (ASYNCcnt path).  Builtin signature (from
// compiler diagnostic): (int4 addrspace(1)*, int4 addrspace(3)*, imm, imm).
// ---------------------------------------------------------------------------
#if __has_builtin(__builtin_amdgcn_global_load_async_to_lds_b128)
#define HAVE_ASYNC_LDS 1
__device__ __forceinline__ void async_g2l_b128(const void* g, void* l) {
    __builtin_amdgcn_global_load_async_to_lds_b128(
        (__attribute__((address_space(1))) vs4i*)g,
        (__attribute__((address_space(3))) vs4i*)l,
        0, 0);
}
__device__ __forceinline__ void async_wait_all() {
#if __has_builtin(__builtin_amdgcn_s_wait_asynccnt)
    __builtin_amdgcn_s_wait_asynccnt(0);
#else
    asm volatile("s_wait_asynccnt 0" ::: "memory");
#endif
}
#endif

// ---------------------------------------------------------------------------
// Stage 0: init workspace (harness poisons ws once; we must re-init per call)
// ---------------------------------------------------------------------------
__global__ void init_ws_kernel(int* __restrict__ cnt, int* __restrict__ minm,
                               int* __restrict__ colcnt) {
    int t = blockIdx.x * blockDim.x + threadIdx.x;
    if (t < BB * NQ) { cnt[t] = 0; minm[t] = IMAX; }
    if (t < BB * MG) { colcnt[t] = 0; }
}

// ---------------------------------------------------------------------------
// Stage 1: dense cost matrix.  pos = sigmoid(logits) @ onehot(labels) via WMMA
// (16x16x32 f16); one-hot staged once per block in LDS so the B operand is a
// single 32B ds load; focal / L1 / GIoU epilogue per element (rcp-based).
// Block = 128 threads = 4 waves; each wave owns 16 queries x all 256 gts.
// ---------------------------------------------------------------------------
__global__ void cost_kernel(const float* __restrict__ logits,
                            const float* __restrict__ pboxes,
                            const float* __restrict__ gboxes,
                            const int*   __restrict__ labels,
                            const float* __restrict__ img,
                            const float* __restrict__ imgt,
                            float* __restrict__ cost) {
    __shared__ _Float16 sP[64 * CPAD];                                  // probs f16
    __shared__ _Float16 sOH[MG * CPAD] __attribute__((aligned(32)));    // one-hot, stored [m][k]
    __shared__ float    sQ[64 * 4] __attribute__((aligned(16)));        // query boxes
    __shared__ float    sG[MG * 4] __attribute__((aligned(16)));        // gt boxes
    __shared__ float    sI[4];                                          // image_size_xyxy[b]
    __shared__ float    sIT[4];                                         // image_size_xyxy_tgt[b]

    const int b   = blockIdx.y;
    const int q0  = blockIdx.x * 64;
    const int tid = threadIdx.x;

#ifdef HAVE_ASYNC_LDS
    // Async DMA of box data into LDS (tracked by ASYNCcnt) while we compute probs.
    for (int i = tid; i < MG; i += 128)       // 256 x 16B gt boxes
        async_g2l_b128(gboxes + (size_t)b * MG * 4 + i * 4, &sG[i * 4]);
    for (int i = tid; i < 64; i += 128)       // 64 x 16B query boxes
        async_g2l_b128(pboxes + ((size_t)(b * NQ + q0)) * 4 + i * 4, &sQ[i * 4]);
#else
    for (int i = tid; i < MG * 4; i += 128)
        sG[i] = gboxes[(size_t)b * MG * 4 + i];
    for (int i = tid; i < 64 * 4; i += 128)
        sQ[i] = pboxes[((size_t)(b * NQ + q0)) * 4 + i];
#endif
    __builtin_prefetch(logits + ((size_t)(b * NQ + q0)) * CC, 0, 0);    // global_prefetch_b8

    // sigmoid(logits) -> f16 probs (zero-padded K to 96 to avoid NaN*0)
    for (int i = tid; i < 64 * CC; i += 128) {
        int q = i / CC, cl = i % CC;
        float x = logits[((size_t)(b * NQ + q0 + q)) * CC + cl];
        float p = fastrcp(1.0f + __expf(-x));
        sP[q * CPAD + cl] = (_Float16)p;
    }
    for (int i = tid; i < 64 * (CPAD - CC); i += 128) {
        int q = i / (CPAD - CC), cl = CC + i % (CPAD - CC);
        sP[q * CPAD + cl] = (_Float16)0.0f;
    }
    // zero one-hot (vectorized, 16B stores)
    {
        uint4* z = (uint4*)sOH;
        for (int i = tid; i < (MG * CPAD * 2) / 16; i += 128)
            z[i] = uint4{0u, 0u, 0u, 0u};
    }
    if (tid < 4) { sI[tid] = img[b * 4 + tid]; sIT[tid] = imgt[b * 4 + tid]; }
#ifdef HAVE_ASYNC_LDS
    async_wait_all();
#endif
    __syncthreads();

    // scatter ones: sOH[m][labels[m]] = 1
    for (int i = tid; i < MG; i += 128)
        sOH[i * CPAD + labels[b * MG + i]] = (_Float16)1.0f;
    __syncthreads();

    const int lane = tid & 31;
    const int w    = tid >> 5;
    const int h    = lane >> 4;     // half-wave select
    const int l16  = lane & 15;
    const int qA   = w * 16 + l16;  // A-matrix row for this lane

    // per-thread reciprocals of the image sizes (hoisted out of all loops)
    const float rI0 = fastrcp(sI[0]),  rI1 = fastrcp(sI[1]);
    const float rI2 = fastrcp(sI[2]),  rI3 = fastrcp(sI[3]);
    const float rT0 = fastrcp(sIT[0]), rT1 = fastrcp(sIT[1]);
    const float rT2 = fastrcp(sIT[2]), rT3 = fastrcp(sIT[3]);

    // A tiles for K chunks 0..2 (contiguous K order matching B: e -> K=16h+e)
    v16h A[3];
    #pragma unroll
    for (int c = 0; c < 3; ++c) {
        #pragma unroll
        for (int e = 0; e < 16; ++e) {
            int k = 32 * c + 16 * h + e;
            A[c][e] = sP[qA * CPAD + k];
        }
    }

    for (int t = 0; t < MG / 16; ++t) {
        const int g = t * 16 + l16;     // B/C column (gt index) for this lane

        // B operand straight from LDS: element e -> K = 32c + 16h + e, contiguous.
        v8f acc = {0.f, 0.f, 0.f, 0.f, 0.f, 0.f, 0.f, 0.f};
        #pragma unroll
        for (int c = 0; c < 3; ++c) {
            v16h Bv = *(const v16h*)&sOH[g * CPAD + 32 * c + 16 * h];
            acc = __builtin_amdgcn_wmma_f32_16x16x32_f16(
                false, A[c], false, Bv, (short)0, acc, false, false);
        }

        // Epilogue: per C/D element r -> query row r+8h, gt col l16.
        const float gx0 = sG[g * 4 + 0], gy0 = sG[g * 4 + 1];
        const float gx1 = sG[g * 4 + 2], gy1 = sG[g * 4 + 3];
        const float ag  = (gx1 - gx0) * (gy1 - gy0);
        const float gn0 = gx0 * rT0, gn1 = gy0 * rT1;
        const float gn2 = gx1 * rT2, gn3 = gy1 * rT3;

        #pragma unroll
        for (int r = 0; r < 8; ++r) {
            int qloc = w * 16 + r + 8 * h;
            float px0 = sQ[qloc * 4 + 0], py0 = sQ[qloc * 4 + 1];
            float px1 = sQ[qloc * 4 + 2], py1 = sQ[qloc * 4 + 3];
            float ap  = (px1 - px0) * (py1 - py0);
            // IoU
            float iw = fmaxf(fminf(px1, gx1) - fmaxf(px0, gx0), 0.f);
            float ih = fmaxf(fminf(py1, gy1) - fmaxf(py0, gy0), 0.f);
            float inter = iw * ih;
            float uni   = ap + ag - inter;
            float iou   = inter * fastrcp(uni);
            // enclosing box
            float ew = fmaxf(fmaxf(px1, gx1) - fminf(px0, gx0), 0.f);
            float eh = fmaxf(fmaxf(py1, gy1) - fminf(py0, gy0), 0.f);
            float enc = ew * eh;
            float cgiou = 1.f - (iou - (enc - uni) * fastrcp(enc));
            // L1 of normalized boxes
            float cb = fabsf(px0 * rI0 - gn0) + fabsf(py0 * rI1 - gn1) +
                       fabsf(px1 * rI2 - gn2) + fabsf(py1 * rI3 - gn3);
            // focal class cost from WMMA-gathered pos
            float pos = acc[r];
            float neg = 1.f - pos;
            float ccl = 0.25f * neg * neg * (-__logf(pos + 1e-8f)) -
                        0.75f * pos * pos * (-__logf(neg + 1e-8f));
            float cst = ccl + 5.0f * cb + 2.0f * cgiou;
            cost[((size_t)(b * NQ + q0 + qloc)) * MG + g] = cst;
        }
    }
}

// ---------------------------------------------------------------------------
// Stage 2: per-gt dynamic-k selection.  One wave32 per (b, m).
// ---------------------------------------------------------------------------
__global__ void topk_kernel(const float* __restrict__ pboxes,
                            const float* __restrict__ gboxes,
                            const float* __restrict__ cost,
                            int* __restrict__ cnt, int* __restrict__ minm) {
    const int m = blockIdx.x, b = blockIdx.y, lane = threadIdx.x;
    const float* gb = gboxes + ((size_t)(b * MG + m)) * 4;
    const float gx0 = gb[0], gy0 = gb[1], gx1 = gb[2], gy1 = gb[3];
    const float ag  = (gx1 - gx0) * (gy1 - gy0);

    float topI[KOTA];
    float topCv[KOTA];
    int   topCi[KOTA];
    #pragma unroll
    for (int j = 0; j < KOTA; ++j) { topI[j] = -3e38f; topCv[j] = 3e38f; topCi[j] = -1; }

    for (int n = lane; n < NQ; n += 32) {
        const float* pb = pboxes + ((size_t)(b * NQ + n)) * 4;
        float px0 = pb[0], py0 = pb[1], px1 = pb[2], py1 = pb[3];
        float ap  = (px1 - px0) * (py1 - py0);
        float iw  = fmaxf(fminf(px1, gx1) - fmaxf(px0, gx0), 0.f);
        float ih  = fmaxf(fminf(py1, gy1) - fmaxf(py0, gy0), 0.f);
        float inter = iw * ih;
        float iou = inter * fastrcp(ap + ag - inter);
        if (iou > topI[KOTA - 1]) {                 // insert, sorted descending
            int j = KOTA - 1;
            while (j > 0 && topI[j - 1] < iou) { topI[j] = topI[j - 1]; --j; }
            topI[j] = iou;
        }
        float cv = cost[((size_t)(b * NQ + n)) * MG + m];
        if (cv < topCv[KOTA - 1]) {                 // insert, sorted ascending
            int j = KOTA - 1;
            while (j > 0 && topCv[j - 1] > cv) {
                topCv[j] = topCv[j - 1]; topCi[j] = topCi[j - 1]; --j;
            }
            topCv[j] = cv; topCi[j] = n;
        }
    }

    // dyn_k = max(trunc(sum of global top-10 IoU), 1)
    float s = 0.f;
    for (int j = 0; j < KOTA; ++j) {
        float v = topI[0];
        float vm = v;
        for (int off = 16; off > 0; off >>= 1) vm = fmaxf(vm, __shfl_xor(vm, off, 32));
        unsigned long long msk = __ballot(v == vm);
        int winner = __ffsll(msk) - 1;
        if (lane == winner) {
            #pragma unroll
            for (int k = 0; k < KOTA - 1; ++k) topI[k] = topI[k + 1];
            topI[KOTA - 1] = -3e38f;
        }
        s += vm;
    }
    int dynk = (int)s;
    if (dynk < 1) dynk = 1;
    if (dynk > KOTA) dynk = KOTA;

    // top-dynk lowest-cost queries -> record assignment
    for (int j = 0; j < KOTA; ++j) {
        float bv = topCv[0]; int bn = topCi[0];
        for (int off = 16; off > 0; off >>= 1) {
            float ov = __shfl_xor(bv, off, 32);
            int   on = __shfl_xor(bn, off, 32);
            if (ov < bv || (ov == bv && on < bn)) { bv = ov; bn = on; }
        }
        if (j < dynk && lane == 0 && bn >= 0) {
            atomicAdd(&cnt[b * NQ + bn], 1);
            atomicMin(&minm[b * NQ + bn], m);
        }
        if (topCi[0] == bn) {                       // winner pops its head
            #pragma unroll
            for (int k = 0; k < KOTA - 1; ++k) { topCv[k] = topCv[k + 1]; topCi[k] = topCi[k + 1]; }
            topCv[KOTA - 1] = 3e38f; topCi[KOTA - 1] = -1;
        }
    }
}

// ---------------------------------------------------------------------------
// Stage 3a: per-query conflict resolution + post-fix column counts.
// ---------------------------------------------------------------------------
__global__ void resolve_kernel(const float* __restrict__ cost,
                               const int* __restrict__ cnt,
                               const int* __restrict__ minm,
                               int* __restrict__ resultm,
                               int* __restrict__ colcnt) {
    int t = blockIdx.x * blockDim.x + threadIdx.x;
    if (t >= BB * NQ) return;
    int c = cnt[t];
    int r;
    if (c == 0) {
        r = IMAX;
    } else if (c == 1) {
        r = minm[t];
    } else {   // multi-assigned: argmin over this query's cost row (first min)
        const float* row = cost + (size_t)t * MG;
        float bv = row[0]; int bi = 0;
        for (int m = 1; m < MG; ++m) {
            float v = row[m];
            if (v < bv) { bv = v; bi = m; }
        }
        r = bi;
    }
    resultm[t] = r;
    if (r != IMAX) atomicAdd(&colcnt[(t / NQ) * MG + r], 1);
}

// ---------------------------------------------------------------------------
// Stage 3b: unmatched gt columns -> argmin query over column.
// ---------------------------------------------------------------------------
__global__ void unmatched_kernel(const float* __restrict__ cost,
                                 const int* __restrict__ colcnt,
                                 int* __restrict__ resultm) {
    const int m = blockIdx.x, b = blockIdx.y, lane = threadIdx.x;
    if (colcnt[b * MG + m] != 0) return;   // uniform per block
    float bv = 3e38f; int bn = IMAX;
    for (int n = lane; n < NQ; n += 32) {
        float v = cost[((size_t)(b * NQ + n)) * MG + m];
        if (v < bv) { bv = v; bn = n; }
    }
    for (int off = 16; off > 0; off >>= 1) {
        float ov = __shfl_xor(bv, off, 32);
        int   on = __shfl_xor(bn, off, 32);
        if (ov < bv || (ov == bv && on < bn)) { bv = ov; bn = on; }
    }
    if (lane == 0) atomicMin(&resultm[b * NQ + bn], m);
}

// ---------------------------------------------------------------------------
// Stage 4: emit sel / gt_idx as floats.
// ---------------------------------------------------------------------------
__global__ void finalize_kernel(const int* __restrict__ resultm,
                                float* __restrict__ outSel,
                                float* __restrict__ outGt) {
    int t = blockIdx.x * blockDim.x + threadIdx.x;
    if (t >= BB * NQ) return;
    int r = resultm[t];
    bool sel = (r != IMAX);
    outSel[t] = sel ? 1.0f : 0.0f;
    outGt[t]  = sel ? (float)r : 0.0f;   // argmax of all-zero row == 0
}

// ---------------------------------------------------------------------------
extern "C" void kernel_launch(void* const* d_in, const int* in_sizes, int n_in,
                              void* d_out, int out_size, void* d_ws, size_t ws_size,
                              hipStream_t stream) {
    (void)in_sizes; (void)n_in; (void)out_size; (void)ws_size;
    const float* logits = (const float*)d_in[0];
    const float* pboxes = (const float*)d_in[1];
    const float* gboxes = (const float*)d_in[2];
    const int*   labels = (const int*)d_in[3];
    const float* img    = (const float*)d_in[4];
    const float* imgt   = (const float*)d_in[5];

    float* out     = (float*)d_out;
    float* outSel  = out;                              // B*N
    float* outGt   = out + (size_t)BB * NQ;            // B*N
    float* outCost = out + (size_t)2 * BB * NQ;        // B*N*M

    int* cnt     = (int*)d_ws;                 // B*N
    int* minm    = cnt + BB * NQ;              // B*N
    int* resultm = minm + BB * NQ;             // B*N
    int* colcnt  = resultm + BB * NQ;          // B*M   (total ~803 KB)

    const int bnBlocks = (BB * NQ + 255) / 256;
    init_ws_kernel<<<dim3(bnBlocks), 256, 0, stream>>>(cnt, minm, colcnt);
    cost_kernel<<<dim3(NQ / 64, BB), 128, 0, stream>>>(logits, pboxes, gboxes,
                                                       labels, img, imgt, outCost);
    topk_kernel<<<dim3(MG, BB), 32, 0, stream>>>(pboxes, gboxes, outCost, cnt, minm);
    resolve_kernel<<<dim3(bnBlocks), 256, 0, stream>>>(outCost, cnt, minm, resultm, colcnt);
    unmatched_kernel<<<dim3(MG, BB), 32, 0, stream>>>(outCost, colcnt, resultm);
    finalize_kernel<<<dim3(bnBlocks), 256, 0, stream>>>(resultm, outSel, outGt);
}
